// TimeSeriesTransformer_78812649881817
// MI455X (gfx1250) — matrix operand order repaired and checked by reading, not compiled, not run on hardware
//
#include <hip/hip_runtime.h>

typedef __bf16 bf16;
typedef bf16 v16bf __attribute__((ext_vector_type(16)));
typedef bf16 v8bf  __attribute__((ext_vector_type(8)));
typedef bf16 v4bf  __attribute__((ext_vector_type(4)));
typedef float v8f  __attribute__((ext_vector_type(8)));
typedef unsigned int u32x4 __attribute__((ext_vector_type(4)));
typedef int i32x8 __attribute__((ext_vector_type(8)));
typedef int i32x4 __attribute__((ext_vector_type(4)));

#if __has_builtin(__builtin_amdgcn_tensor_load_to_lds)
#define HAVE_TDM 1
#else
#define HAVE_TDM 0
#endif

#define DEV __device__ __forceinline__

DEV v8f wmma_bf(v16bf a, v16bf b, v8f c) {
  return __builtin_amdgcn_wmma_f32_16x16x32_bf16(false, a, false, b, (short)0, c, false, false);
}
DEV v8f zero8() { v8f z; for (int i = 0; i < 8; i++) z[i] = 0.f; return z; }
DEV v16bf zero16() { v16bf z; for (int i = 0; i < 16; i++) z[i] = (bf16)0.f; return z; }
// A operand: elems 0..7 <- 16B at lo, elems 8..15 <- 16B at hi
DEV v16bf ldA(const bf16* lo, const bf16* hi) {
  union { v16bf v; v8bf h[2]; } u;
  u.h[0] = *(const v8bf*)lo; u.h[1] = *(const v8bf*)hi; return u.v;
}
// B operand: 32 contiguous bytes
DEV v16bf ldB(const bf16* p) {
  union { v16bf v; v8bf h[2]; } u;
  u.h[0] = *(const v8bf*)p; u.h[1] = *(const v8bf*)(p + 8); return u.v;
}
// A operand from LDS row (8-byte chunks, row may be 8B-aligned only)
DEV v16bf ldA4(const bf16* row, int half) {
  union { v16bf v; v4bf q[4]; } u;
  u.q[0] = *(const v4bf*)(row + 16 * half);
  u.q[1] = *(const v4bf*)(row + 16 * half + 4);
  u.q[2] = *(const v4bf*)(row + 16 * half + 8);
  u.q[3] = *(const v4bf*)(row + 16 * half + 12);
  return u.v;
}

// ---------------- conversion / prep kernels ----------------
__global__ void cvt_bf16(const float* in, bf16* out, int n) {
  int i = blockIdx.x * blockDim.x + threadIdx.x;
  if (i < n) out[i] = (bf16)in[i];
}
// w [Cout,Cin,K] f32 -> wt [Cout,K,Cin] bf16
__global__ void cvt_convw(const float* w, bf16* wt, int Cout, int Cin, int K) {
  int i = blockIdx.x * blockDim.x + threadIdx.x;
  int total = Cout * Cin * K;
  if (i >= total) return;
  int k = i % K; int t = i / K; int ci = t % Cin; int co = t / Cin;
  wt[(co * K + k) * Cin + ci] = (bf16)w[i];
}
// fold conv-bias + BN into scale/shift
__global__ void bn_fold(const float* cb, const float* g, const float* bb,
                        const float* m, const float* v, float* sc, float* sh, int C) {
  int c = blockIdx.x * blockDim.x + threadIdx.x;
  if (c >= C) return;
  float s = g[c] * rsqrtf(v[c] + 1e-5f);
  sc[c] = s;
  sh[c] = s * (cb[c] - m[c]) + bb[c];
}

// ---------------- CNN kernels ----------------
__global__ void in_conv_kernel(const float* x, const float* w,
                               const float* sc, const float* sh, bf16* out) {
  int i = blockIdx.x * blockDim.x + threadIdx.x;
  if (i >= 32 * 512 * 32) return;
  int co = i & 31; int l = (i >> 5) & 511; int n = i >> 14;
  float s = 0.f;
  for (int k = 0; k < 15; k++) {
    int ls = l + k - 7;
    if (ls < 0 || ls >= 512) continue;
    const float* xr = x + (size_t)(n * 512 + ls) * 6;
    for (int ci = 0; ci < 6; ci++) s += xr[ci] * w[(co * 6 + ci) * 15 + k];
  }
  float y = s * sc[co] + sh[co];
  y = fmaxf(y, 0.f);
  out[(size_t)(n * 512 + l) * 32 + co] = (bf16)y;
}

// channels-last implicit-GEMM conv: X [32,512,Cin] bf16, Wt [Cout,K,Cin] bf16
template <int NT>
__global__ void conv_gemm(const bf16* X, const bf16* Wt, const float* sc, const float* sh,
                          bf16* out, int Cin, int Kw, int pad, int Cout,
                          int outStride, int coOff, int relu) {
  const int L = 512;
  int lane = threadIdx.x & 31;
  int wav = blockIdx.x * (blockDim.x >> 5) + (threadIdx.x >> 5);
  int mTiles = (32 * L) / 16;
  int nSlabs = Cout / (16 * NT);
  if (wav >= mTiles * nSlabs) return;
  int mTile = wav % mTiles, nSlab = wav / mTiles;
  int row0 = mTile * 16;
  int n = row0 / L, l0 = row0 % L;
  int half = lane >> 4, lid = lane & 15;
  int myl = l0 + lid;
  int co0 = nSlab * NT * 16;
  v8f acc[NT];
  for (int t = 0; t < NT; t++) acc[t] = zero8();
  for (int k = 0; k < Kw; k++) {
    int le = myl + k - pad;
    bool valid = (le >= 0 && le < L);
    const bf16* arow = X + (size_t)(n * L + le) * Cin;
    for (int cc = 0; cc < Cin; cc += 32) {
      v16bf a = zero16();
      if (valid) a = ldA(arow + cc + 16 * half, arow + cc + 16 * half + 8);
#pragma unroll
      for (int t = 0; t < NT; t++) {
        int c = co0 + t * 16 + lid;
        v16bf b = ldB(Wt + ((size_t)c * Kw + k) * Cin + cc + 16 * half);
        acc[t] = wmma_bf(a, b, acc[t]);
      }
    }
  }
#pragma unroll
  for (int t = 0; t < NT; t++) {
    int c = co0 + t * 16 + lid;
    float s = sc[c], h = sh[c];
#pragma unroll
    for (int r = 0; r < 8; r++) {
      int m = r + 8 * half;
      float y = acc[t][r] * s + h;
      if (relu) y = fmaxf(y, 0.f);
      out[(size_t)(row0 + m) * outStride + coOff + c] = (bf16)y;
    }
  }
}

__global__ void maxpool3(const bf16* in, bf16* out, int C) {
  int i = blockIdx.x * blockDim.x + threadIdx.x;
  if (i >= 32 * 512 * C) return;
  int c = i % C; int l = (i / C) % 512; int n = i / (C * 512);
  float m = (float)in[(size_t)(n * 512 + l) * C + c];
  if (l > 0)   m = fmaxf(m, (float)in[(size_t)(n * 512 + l - 1) * C + c]);
  if (l < 511) m = fmaxf(m, (float)in[(size_t)(n * 512 + l + 1) * C + c]);
  out[(size_t)(n * 512 + l) * C + c] = (bf16)m;
}

__global__ void add_relu(const bf16* a, const bf16* b, bf16* o, int n) {
  int i = blockIdx.x * blockDim.x + threadIdx.x;
  if (i < n) o[i] = (bf16)fmaxf((float)a[i] + (float)b[i], 0.f);
}

// ---------------- small linear GEMM (K=64 QKV):  out = A[M,64] * W[64,64]^T ----------------
__global__ void lin_gemm(const bf16* A, const bf16* W, const float* bias,
                         float* outF, bf16* outB, const float* pe,
                         int M, int N, int K, int relu) {
  int lane = threadIdx.x & 31;
  int wav = blockIdx.x * (blockDim.x >> 5) + (threadIdx.x >> 5);
  int mTiles = M / 16, nSlabs = N / 64;
  if (wav >= mTiles * nSlabs) return;
  int mTile = wav % mTiles, nSlab = wav / mTiles;
  int row0 = mTile * 16;
  int half = lane >> 4, lid = lane & 15;
  const bf16* arow = A + (size_t)(row0 + lid) * K;
  int co0 = nSlab * 64;
  v8f acc[4];
  for (int t = 0; t < 4; t++) acc[t] = zero8();
  for (int kk = 0; kk < K; kk += 32) {
    v16bf a = ldA(arow + kk + 16 * half, arow + kk + 16 * half + 8);
#pragma unroll
    for (int t = 0; t < 4; t++) {
      v16bf b = ldB(W + (size_t)(co0 + t * 16 + lid) * K + kk + 16 * half);
      acc[t] = wmma_bf(a, b, acc[t]);
    }
  }
#pragma unroll
  for (int t = 0; t < 4; t++) {
    int c = co0 + t * 16 + lid;
    float bs = bias ? bias[c] : 0.f;
#pragma unroll
    for (int r = 0; r < 8; r++) {
      int m = row0 + r + 8 * half;
      float y = acc[t][r] + bs;
      if (pe) y += pe[(size_t)(m & 511) * N + c];
      if (relu) y = fmaxf(y, 0.f);
      if (outF) outF[(size_t)m * N + c] = y;
      if (outB) outB[(size_t)m * N + c] = (bf16)y;
    }
  }
}

// ---------------- big linear GEMM with TDM-staged B in LDS ----------------
// Block = 8 waves = 128 rows x 64 cols. B staged in 256-col chunks:
// LDS row layout: 64 rows, 512B payload + 16B pad => 528B stride (conflict-free b128 reads).
#define KC 256
#define BROW 528
__global__ void __launch_bounds__(256)
lin_gemm_lds(const bf16* A, const bf16* W, const float* bias,
             float* outF, bf16* outB, const float* pe,
             int M, int N, int K, int relu) {
  extern __shared__ char Bs[];
  int tid = threadIdx.x;
  int lane = tid & 31;
  int wib = tid >> 5;
  int mBlocks = M / 128;
  int mBlk = blockIdx.x % mBlocks, nSlab = blockIdx.x / mBlocks;
  int co0 = nSlab * 64;
  int half = lane >> 4, lid = lane & 15;
  int row0 = (mBlk * 8 + wib) * 16;
  const bf16* arow = A + (size_t)(row0 + lid) * K;

  v8f acc[4];
  for (int t = 0; t < 4; t++) acc[t] = zero8();
  int rB[4];
#pragma unroll
  for (int t = 0; t < 4; t++) rB[t] = (t * 16 + lid) * BROW;

  for (int kc = 0; kc < K; kc += KC) {
#if HAVE_TDM
    if (wib == 0) {
      // Tensor DMA descriptor (ISA ch.8): 2D tile 256x64 bf16, row stride K,
      // LDS padding: 16B after every 128 DWORDs (one 512B row).
      unsigned ldsAddr = (unsigned)(size_t)(void*)Bs;
      unsigned long long ga =
          (unsigned long long)(size_t)(W + (size_t)co0 * K + kc);
      u32x4 g0; i32x8 g1; i32x4 g2, g3; i32x8 g4;
      g0[0] = 1u;                                   // count=1, user mode
      g0[1] = ldsAddr;                              // lds_addr
      g0[2] = (unsigned)(ga & 0xffffffffu);         // global_addr lo
      g0[3] = (unsigned)((ga >> 32) & 0x1ffffffu) | (2u << 30);  // hi | type=2
      g1[0] = (int)((1u << 16) |                    // data_size = 2B
                    (1u << 20) |                    // pad_enable
                    (6u << 22) |                    // pad_interval = 128 DW (512B)
                    (3u << 25));                    // pad_amount = 4 DW (16B)
      g1[1] = (int)((KC & 0xffffu) << 16);          // tensor_dim0 lo16 (=256)
      g1[2] = (int)((64u & 0xffffu) << 16);         // dim0 hi16=0 | tensor_dim1 lo16 (=64)
      g1[3] = (int)((KC & 0xffffu) << 16);          // dim1 hi16=0 | tile_dim0 (=256)
      g1[4] = 64;                                   // tile_dim1=64, tile_dim2=0
      g1[5] = K;                                    // tensor_dim0_stride lo32
      g1[6] = 0; g1[7] = 0;                         // stride hi, dim1_stride=0
      g2[0] = 0; g2[1] = 0; g2[2] = 0; g2[3] = 0;
      g3 = g2;
      g4[0] = 0; g4[1] = 0; g4[2] = 0; g4[3] = 0;
      g4[4] = 0; g4[5] = 0; g4[6] = 0; g4[7] = 0;
      __builtin_amdgcn_tensor_load_to_lds(g0, g1, g2, g3, g4, 0);
      __builtin_amdgcn_s_wait_tensorcnt(0);
    }
#else
    for (int idx = tid; idx < 64 * (KC / 8); idx += 256) {
      int r = idx >> 5, j = idx & 31;
      const bf16* src = W + (size_t)(co0 + r) * K + kc + j * 8;
      *(v8bf*)(Bs + r * BROW + j * 16) = *(const v8bf*)src;
    }
#endif
    __syncthreads();
    // A operand for first sub-step of this chunk
    v16bf a = ldA(arow + kc + 16 * half, arow + kc + 16 * half + 8);
#pragma unroll
    for (int kk = 0; kk < KC; kk += 32) {
      v16bf an = a;
      if (kc + kk + 32 < K)  // prefetch next A one step ahead
        an = ldA(arow + kc + kk + 32 + 16 * half, arow + kc + kk + 40 + 16 * half);
      int koff = (kk + 16 * half) * 2;
#pragma unroll
      for (int t = 0; t < 4; t++) {
        v16bf b = ldB((const bf16*)(Bs + rB[t] + koff));
        acc[t] = wmma_bf(a, b, acc[t]);
      }
      a = an;
    }
    __syncthreads();
  }
#pragma unroll
  for (int t = 0; t < 4; t++) {
    int c = co0 + t * 16 + lid;
    float bs = bias ? bias[c] : 0.f;
#pragma unroll
    for (int r = 0; r < 8; r++) {
      int m = row0 + r + 8 * half;
      float y = acc[t][r] + bs;
      if (pe) y += pe[(size_t)(m & 511) * N + c];
      if (relu) y = fmaxf(y, 0.f);
      if (outF) outF[(size_t)m * N + c] = y;
      if (outB) outB[(size_t)m * N + c] = (bf16)y;
    }
  }
}

// ---------------- transformer helper kernels ----------------
__global__ void vtrans(const bf16* v, bf16* vt) {
  int i = blockIdx.x * blockDim.x + threadIdx.x;
  if (i >= 32 * 512 * 512) return;
  int e = i & 511; int l = (i >> 9) & 511; int n = i >> 18;
  vt[((size_t)(n * 512 + e)) * 512 + l] = v[i];
}

DEV float redmax16(float v) {
  v = fmaxf(v, __shfl_xor(v, 1, 32));
  v = fmaxf(v, __shfl_xor(v, 2, 32));
  v = fmaxf(v, __shfl_xor(v, 4, 32));
  v = fmaxf(v, __shfl_xor(v, 8, 32));
  return v;
}
DEV float redsum16(float v) {
  v += __shfl_xor(v, 1, 32); v += __shfl_xor(v, 2, 32);
  v += __shfl_xor(v, 4, 32); v += __shfl_xor(v, 8, 32);
  return v;
}

// fused flash attention: one wave = (n, head, 16 q rows)
__global__ void __launch_bounds__(128)
attn_kernel(const bf16* Q, const bf16* K, const bf16* Vt, bf16* O) {
  int lane = threadIdx.x & 31;
  int wib = threadIdx.x >> 5;
  int wav = blockIdx.x * 4 + wib;
  int qT = wav & 31; int head = (wav >> 5) & 7; int n = wav >> 8;
  int half = lane >> 4, lid = lane & 15;
  const float scale = 0.044194173824159216f; // 1/sqrt(512)

  const bf16* qrow = Q + ((size_t)((n * 512 + qT * 16 + lid) * 8 + head)) * 64;
  v16bf qa0 = ldA(qrow + 16 * half, qrow + 16 * half + 8);
  v16bf qa1 = ldA(qrow + 32 + 16 * half, qrow + 40 + 16 * half);

  v8f o0 = zero8(), o1 = zero8(), o2 = zero8(), o3 = zero8();
  float mrun[8], lrun[8];
  for (int r = 0; r < 8; r++) { mrun[r] = -1e30f; lrun[r] = 0.f; }

  __shared__ bf16 smem[4 * 16 * 40];
  bf16* sp = smem + wib * 16 * 40;

  const bf16* kbase = K + ((size_t)(n * 512 * 8 + head)) * 64;
  const bf16* vbase = Vt + ((size_t)(n * 8 + head)) * 64 * 512;

  for (int p = 0; p < 16; p++) {
    const bf16* kr0 = kbase + (size_t)(p * 32 + lid) * 512;
    const bf16* kr1 = kr0 + (size_t)16 * 512;
    v8f sa = zero8(), sb = zero8();
    sa = wmma_bf(qa0, ldB(kr0 + 16 * half), sa);
    sa = wmma_bf(qa1, ldB(kr0 + 32 + 16 * half), sa);
    sb = wmma_bf(qa0, ldB(kr1 + 16 * half), sb);
    sb = wmma_bf(qa1, ldB(kr1 + 32 + 16 * half), sb);

    float corr[8];
#pragma unroll
    for (int r = 0; r < 8; r++) {
      float xa = sa[r] * scale, xb = sb[r] * scale;
      float mx = redmax16(fmaxf(xa, xb));
      float nm = fmaxf(mrun[r], mx);
      float pa = __expf(xa - nm), pb = __expf(xb - nm);
      float rs = redsum16(pa + pb);
      float c = __expf(mrun[r] - nm);
      lrun[r] = lrun[r] * c + rs;
      mrun[r] = nm;
      corr[r] = c;
      sp[(r + 8 * half) * 40 + lid] = (bf16)pa;
      sp[(r + 8 * half) * 40 + 16 + lid] = (bf16)pb;
    }
#pragma unroll
    for (int r = 0; r < 8; r++) {
      o0[r] *= corr[r]; o1[r] *= corr[r]; o2[r] *= corr[r]; o3[r] *= corr[r];
    }
    v16bf pv = ldA4(sp + lid * 40, half);
    const bf16* vb = vbase + (size_t)lid * 512 + p * 32 + 16 * half;
    o0 = wmma_bf(pv, ldB(vb), o0);
    o1 = wmma_bf(pv, ldB(vb + (size_t)16 * 512), o1);
    o2 = wmma_bf(pv, ldB(vb + (size_t)32 * 512), o2);
    o3 = wmma_bf(pv, ldB(vb + (size_t)48 * 512), o3);
  }
#pragma unroll
  for (int r = 0; r < 8; r++) {
    float inv = 1.f / lrun[r];
    size_t row = (size_t)(n * 512 + qT * 16 + r + 8 * half) * 512 + head * 64 + lid;
    O[row +  0] = (bf16)(o0[r] * inv);
    O[row + 16] = (bf16)(o1[r] * inv);
    O[row + 32] = (bf16)(o2[r] * inv);
    O[row + 48] = (bf16)(o3[r] * inv);
  }
}

// LayerNorm(o + res): one wave per 512-wide row
__global__ void ln_kernel(const float* o, const float* res, const float* g,
                          const float* b, float* outF, bf16* outB) {
  int lane = threadIdx.x & 31;
  int row = blockIdx.x * (blockDim.x >> 5) + (threadIdx.x >> 5);
  if (row >= 16384) return;
  const float* orow = o + (size_t)row * 512;
  const float* rrow = res + (size_t)row * 512;
  float vals[16]; float s = 0.f;
#pragma unroll
  for (int j = 0; j < 16; j++) { vals[j] = orow[j * 32 + lane] + rrow[j * 32 + lane]; s += vals[j]; }
  for (int m = 1; m < 32; m <<= 1) s += __shfl_xor(s, m, 32);
  float mu = s * (1.f / 512.f);
  float vs = 0.f;
#pragma unroll
  for (int j = 0; j < 16; j++) { float d = vals[j] - mu; vs += d * d; }
  for (int m = 1; m < 32; m <<= 1) vs += __shfl_xor(vs, m, 32);
  float rs = rsqrtf(vs * (1.f / 512.f) + 1e-5f);
#pragma unroll
  for (int j = 0; j < 16; j++) {
    int c = j * 32 + lane;
    float y = g[c] * (vals[j] - mu) * rs + b[c];
    outF[(size_t)row * 512 + c] = y;
    outB[(size_t)row * 512 + c] = (bf16)y;
  }
}

__global__ void meanpool(const float* h, float* out) {
  int i = blockIdx.x * blockDim.x + threadIdx.x;
  if (i >= 32 * 512) return;
  int e = i & 511; int n = i >> 9;
  float s = 0.f;
  for (int l = 0; l < 512; l++) s += h[((size_t)(n * 512 + l)) * 512 + e];
  out[i] = s * (1.f / 512.f);
}

__global__ void dense(const float* in, const float* w, const float* b, float* out,
                      int M, int N, int K, int relu) {
  int i = blockIdx.x * blockDim.x + threadIdx.x;
  if (i >= M * N) return;
  int n = i % N; int m = i / N;
  float s = b[n];
  for (int k = 0; k < K; k++) s += in[(size_t)m * K + k] * w[(size_t)n * K + k];
  if (relu) s = fmaxf(s, 0.f);
  out[i] = s;
}

// ---------------- host orchestration ----------------
// jax tree_leaves order (sorted dict keys at every level)
enum {
  CLS0_B = 0, CLS0_W, CLS1_B, CLS1_W, CLS2_B, CLS2_W,
  FE_B = 6, FE_W = 7,
  IC_B = 8, IC_BNB, IC_BNG, IC_BNM, IC_BNV, IC_W,
  INC0 = 14, INC1 = 50,            // per block: 6 branches x (b,bnb,bng,bnm,bnv,w)
  LAY0 = 86,                       // per layer: ff1b,ff1w,ff2b,ff2w,fob,fow,ln1b,ln1g,ln2b,ln2g,wk,wq,wv
  POS_EMB = 125,
  RES0 = 126, RES1 = 144,          // per block: c1,c2,sk x (b,bnb,bng,bnm,bnv,w)
  X_IN = 162
};

struct CW { bf16* wt; float* sc; float* sh; };

extern "C" void kernel_launch(void* const* d_in, const int* in_sizes, int n_in,
                              void* d_out, int out_size, void* d_ws, size_t ws_size,
                              hipStream_t stream) {
  (void)in_sizes; (void)n_in; (void)out_size; (void)ws_size;
  auto F = [&](int i) { return (const float*)d_in[i]; };
  char* base = (char*)d_ws;
  size_t off = 0;
  auto alloc = [&](size_t bytes) -> void* {
    off = (off + 255) & ~(size_t)255;
    void* p = base + off; off += bytes; return p;
  };
  auto cvt = [&](int idx, size_t n) -> bf16* {
    bf16* p = (bf16*)alloc(n * 2);
    cvt_bf16<<<(int)((n + 255) / 256), 256, 0, stream>>>(F(idx), p, (int)n);
    return p;
  };
  auto prep = [&](int bidx, int Cout, int Cin, int K) -> CW {
    CW c; size_t wn = (size_t)Cout * Cin * K;
    c.wt = (bf16*)alloc(wn * 2);
    c.sc = (float*)alloc((size_t)Cout * 4);
    c.sh = (float*)alloc((size_t)Cout * 4);
    cvt_convw<<<(int)((wn + 255) / 256), 256, 0, stream>>>(F(bidx + 5), c.wt, Cout, Cin, K);
    bn_fold<<<(Cout + 255) / 256, 256, 0, stream>>>(F(bidx), F(bidx + 2), F(bidx + 1),
                                                    F(bidx + 3), F(bidx + 4), c.sc, c.sh, Cout);
    return c;
  };
  auto conv = [&](const CW& c, const bf16* in, bf16* out, int Cin, int K, int pad,
                  int Cout, int outStride, int coOff, int relu) {
    if (Cout % 64 == 0) {
      int waves = 1024 * (Cout / 64);
      conv_gemm<4><<<waves / 8, 256, 0, stream>>>(in, c.wt, c.sc, c.sh, out, Cin, K, pad,
                                                  Cout, outStride, coOff, relu);
    } else {
      int waves = 1024 * (Cout / 32);
      conv_gemm<2><<<waves / 8, 256, 0, stream>>>(in, c.wt, c.sc, c.sh, out, Cin, K, pad,
                                                  Cout, outStride, coOff, relu);
    }
  };
  auto lin = [&](const bf16* A, const bf16* W, const float* bias, float* oF, bf16* oB,
                 const float* pe, int M, int N, int K, int relu) {
    if (K >= 256) {
      int blocks = (M / 128) * (N / 64);
      lin_gemm_lds<<<blocks, 256, 64 * BROW, stream>>>(A, W, bias, oF, oB, pe, M, N, K, relu);
    } else {
      int waves = (M / 16) * (N / 64);
      lin_gemm<<<waves / 8, 256, 0, stream>>>(A, W, bias, oF, oB, pe, M, N, K, relu);
    }
  };

  // ---- weight conversion ----
  bf16* feW = cvt(FE_W, 512 * 512);
  bf16 *qW[3], *kW[3], *vW[3], *foW[3], *f1W[3], *f2W[3];
  for (int j = 0; j < 3; j++) {
    int b = LAY0 + j * 13;
    f1W[j] = cvt(b + 1, 1024 * 512);
    f2W[j] = cvt(b + 3, 512 * 1024);
    foW[j] = cvt(b + 5, 512 * 512);
    kW[j] = cvt(b + 10, 64 * 64);
    qW[j] = cvt(b + 11, 64 * 64);
    vW[j] = cvt(b + 12, 64 * 64);
  }
  float* icSc = (float*)alloc(32 * 4);
  float* icSh = (float*)alloc(32 * 4);
  bn_fold<<<1, 256, 0, stream>>>(F(IC_B), F(IC_BNG), F(IC_BNB), F(IC_BNM), F(IC_BNV), icSc, icSh, 32);

  CW inc[2][6];
  for (int i = 0; i < 2; i++) {
    int ic = (i == 0) ? 32 : 128;
    int b = (i == 0) ? INC0 : INC1;
    inc[i][0] = prep(b + 0, 32, ic, 1);   // b1
    inc[i][1] = prep(b + 6, 32, ic, 1);   // b2a
    inc[i][2] = prep(b + 12, 32, 32, 3);  // b2b
    inc[i][3] = prep(b + 18, 32, ic, 1);  // b3a
    inc[i][4] = prep(b + 24, 32, 32, 5);  // b3b
    inc[i][5] = prep(b + 30, 32, ic, 1);  // b4
  }
  CW res[2][3];
  for (int i = 0; i < 2; i++) {
    int cin = (i == 0) ? 128 : 256, o = cin * 2;
    int b = (i == 0) ? RES0 : RES1;
    res[i][0] = prep(b + 0, o, cin, 3);   // c1
    res[i][1] = prep(b + 6, o, o, 3);     // c2
    res[i][2] = prep(b + 12, o, cin, 1);  // sk
  }

  // ---- activations ----
  const size_t NL = 32 * 512;
  bf16* cnn0 = (bf16*)alloc(NL * 32 * 2);
  bf16* t32a = (bf16*)alloc(NL * 32 * 2);
  bf16* t32b = (bf16*)alloc(NL * 32 * 2);
  bf16* cnnP = (bf16*)alloc(NL * 128 * 2);
  bf16* cnn1 = (bf16*)alloc(NL * 128 * 2);
  bf16* cnn2 = (bf16*)alloc(NL * 128 * 2);
  bf16* r0a = (bf16*)alloc(NL * 256 * 2);
  bf16* r0b = (bf16*)alloc(NL * 256 * 2);
  bf16* r0s = (bf16*)alloc(NL * 256 * 2);
  bf16* r0h = (bf16*)alloc(NL * 256 * 2);
  bf16* r1a = (bf16*)alloc(NL * 512 * 2);
  bf16* r1b = (bf16*)alloc(NL * 512 * 2);
  bf16* r1s = (bf16*)alloc(NL * 512 * 2);
  bf16* hcnn = (bf16*)alloc(NL * 512 * 2);
  bf16* hbf = (bf16*)alloc(NL * 512 * 2);
  float* hf32 = (float*)alloc(NL * 512 * 4);
  bf16* qb = (bf16*)alloc(NL * 512 * 2);
  bf16* kb = (bf16*)alloc(NL * 512 * 2);
  bf16* vb = (bf16*)alloc(NL * 512 * 2);
  bf16* vtb = (bf16*)alloc(NL * 512 * 2);
  bf16* at = (bf16*)alloc(NL * 512 * 2);
  float* of32 = (float*)alloc(NL * 512 * 4);
  float* x1f = (float*)alloc(NL * 512 * 4);
  bf16* x1b = (bf16*)alloc(NL * 512 * 2);
  bf16* ffb = (bf16*)alloc(NL * 1024 * 2);
  float* pooled = (float*)alloc(32 * 512 * 4);
  float* dc1 = (float*)alloc(32 * 512 * 4);
  float* dc2 = (float*)alloc(32 * 256 * 4);

  // ---- CNN ----
  in_conv_kernel<<<(32 * 512 * 32) / 256, 256, 0, stream>>>(F(X_IN), F(IC_W), icSc, icSh, cnn0);

  bf16* bin = cnn0;
  for (int i = 0; i < 2; i++) {
    int ic = (i == 0) ? 32 : 128;
    bf16* bout = (i == 0) ? cnn1 : cnn2;
    conv(inc[i][0], bin, bout, ic, 1, 0, 32, 128, 0, 1);       // b1
    conv(inc[i][1], bin, t32a, ic, 1, 0, 32, 32, 0, 1);        // b2a
    conv(inc[i][2], t32a, bout, 32, 3, 1, 32, 128, 32, 1);     // b2b
    conv(inc[i][3], bin, t32b, ic, 1, 0, 32, 32, 0, 1);        // b3a
    conv(inc[i][4], t32b, bout, 32, 5, 2, 32, 128, 64, 1);     // b3b
    maxpool3<<<(int)((NL * ic + 255) / 256), 256, 0, stream>>>(bin, cnnP, ic);
    conv(inc[i][5], cnnP, bout, ic, 1, 0, 32, 128, 96, 1);     // b4
    bin = bout;
  }
  // residual block 0: 128 -> 256
  conv(res[0][2], cnn2, r0s, 128, 1, 0, 256, 256, 0, 0);
  conv(res[0][0], cnn2, r0a, 128, 3, 1, 256, 256, 0, 1);
  conv(res[0][1], r0a, r0b, 256, 3, 1, 256, 256, 0, 0);
  add_relu<<<(int)((NL * 256 + 255) / 256), 256, 0, stream>>>(r0b, r0s, r0h, (int)(NL * 256));
  // residual block 1: 256 -> 512
  conv(res[1][2], r0h, r1s, 256, 1, 0, 512, 512, 0, 0);
  conv(res[1][0], r0h, r1a, 256, 3, 1, 512, 512, 0, 1);
  conv(res[1][1], r1a, r1b, 512, 3, 1, 512, 512, 0, 0);
  add_relu<<<(int)((NL * 512 + 255) / 256), 256, 0, stream>>>(r1b, r1s, hcnn, (int)(NL * 512));

  // ---- feature embedding + positional ----
  lin(hcnn, feW, F(FE_B), hf32, hbf, F(POS_EMB), 16384, 512, 512, 0);

  // ---- transformer layers ----
  for (int j = 0; j < 3; j++) {
    int b = LAY0 + j * 13;
    lin(hbf, qW[j], nullptr, nullptr, qb, nullptr, 131072, 64, 64, 0);
    lin(hbf, kW[j], nullptr, nullptr, kb, nullptr, 131072, 64, 64, 0);
    lin(hbf, vW[j], nullptr, nullptr, vb, nullptr, 131072, 64, 64, 0);
    vtrans<<<(32 * 512 * 512) / 256, 256, 0, stream>>>(vb, vtb);
    attn_kernel<<<8192 / 4, 128, 0, stream>>>(qb, kb, vtb, at);
    lin(at, foW[j], F(b + 4), of32, nullptr, nullptr, 16384, 512, 512, 0);
    ln_kernel<<<2048, 256, 0, stream>>>(of32, hf32, F(b + 7), F(b + 6), x1f, x1b);
    lin(x1b, f1W[j], F(b + 0), nullptr, ffb, nullptr, 16384, 1024, 512, 1);
    lin(ffb, f2W[j], F(b + 2), of32, nullptr, nullptr, 16384, 512, 1024, 0);
    ln_kernel<<<2048, 256, 0, stream>>>(of32, x1f, F(b + 9), F(b + 8), hf32, hbf);
  }

  // ---- classifier ----
  meanpool<<<64, 256, 0, stream>>>(hf32, pooled);
  dense<<<64, 256, 0, stream>>>(pooled, F(CLS0_W), F(CLS0_B), dc1, 32, 512, 512, 1);
  dense<<<32, 256, 0, stream>>>(dc1, F(CLS1_W), F(CLS1_B), dc2, 32, 256, 512, 1);
  dense<<<2, 256, 0, stream>>>(dc2, F(CLS2_W), F(CLS2_B), (float*)d_out, 32, 10, 256, 0);
}
